// CoupledAttention_36687610642708
// MI455X (gfx1250) — matrix operand order repaired and checked by reading, not compiled, not run on hardware
//
#include <hip/hip_runtime.h>
#include <hip/hip_bf16.h>
#include <math.h>

#define N_   8
#define F_   64
#define P_   25
#define H_   6
#define D_   192
#define HD_  32
#define T_   (F_*P_)          // 1600
#define MAXHOP_ 5
#define SCALE_ 0.1767766952966369f  // 32^-0.5

typedef unsigned short u16;
typedef __attribute__((ext_vector_type(16))) __bf16 v16bf;
typedef __attribute__((ext_vector_type(8)))  float  v8f;

struct alignas(16) U4 { unsigned int a, b, c, d; };

static __device__ __forceinline__ u16 f2bf(float f) {
    unsigned int u = __float_as_uint(f);
    u += 0x7FFFu + ((u >> 16) & 1u);       // round-to-nearest-even
    return (u16)(u >> 16);
}

// A-matrix 16x32 bf16 fragment (ISA 7.12.2): lane<16 holds K=0..7,16..23 of its
// row; lane>=16 holds K=8..15,24..31.  rowbase points at this lane's row.
static __device__ __forceinline__ v16bf load_a_frag(const u16* rowbase, int half) {
    union { U4 q[2]; v16bf v; } u;
    u.q[0] = *reinterpret_cast<const U4*>(rowbase + half * 8);
    u.q[1] = *reinterpret_cast<const U4*>(rowbase + half * 8 + 16);
    return u.v;
}

// B-matrix 32x16 bf16 fragment: lane ln holds column ln; lane<16 holds
// K=0..7,16..23, lane>=16 holds K=8..15,24..31.  colbase points at this lane's column (K-major).
static __device__ __forceinline__ v16bf load_b_frag(const u16* colbase, int half) {
    union { U4 q[2]; v16bf v; } u;
    u.q[0] = *reinterpret_cast<const U4*>(colbase + half * 8);
    u.q[1] = *reinterpret_cast<const U4*>(colbase + half * 8 + 16);
    return u.v;
}

// ---------------------------------------------------------------- conversion
__global__ void f32_to_bf16_kernel(const float* __restrict__ src,
                                   u16* __restrict__ dst, int n) {
    int i = blockIdx.x * blockDim.x + threadIdx.x;
    if (i < n) dst[i] = f2bf(src[i]);
}

// ---------------------------------------------------------------- bias tables
// table_t (127,H) / table_h (5,H) via Linear(1,192)->ReLU->Linear(192,H),
// then expand to bias_f (H,F,F) and bias_p (H,P,P).
__global__ void bias_tables_kernel(const float* __restrict__ t_w1, const float* __restrict__ t_b1,
                                   const float* __restrict__ t_w2, const float* __restrict__ t_b2,
                                   const float* __restrict__ h_w1, const float* __restrict__ h_b1,
                                   const float* __restrict__ h_w2, const float* __restrict__ h_b2,
                                   const int*   __restrict__ hops,
                                   float* __restrict__ bias_f, float* __restrict__ bias_p) {
    __shared__ float tabT[(2 * F_ - 1) * H_];
    __shared__ float tabH[MAXHOP_ * H_];
    int tid = threadIdx.x;
    for (int idx = tid; idx < (2 * F_ - 1 + MAXHOP_) * H_; idx += blockDim.x) {
        if (idx < (2 * F_ - 1) * H_) {
            int i = idx / H_, h = idx % H_;
            float pos = (float)(i - (F_ - 1));
            float acc = t_b2[h];
            for (int j = 0; j < D_; j++) {
                float hv = pos * t_w1[j] + t_b1[j];
                hv = hv > 0.f ? hv : 0.f;
                acc += hv * t_w2[h * D_ + j];
            }
            tabT[idx] = acc;
        } else {
            int rem = idx - (2 * F_ - 1) * H_;
            int i = rem / H_, h = rem % H_;
            float pos = (float)i;
            float acc = h_b2[h];
            for (int j = 0; j < D_; j++) {
                float hv = pos * h_w1[j] + h_b1[j];
                hv = hv > 0.f ? hv : 0.f;
                acc += hv * h_w2[h * D_ + j];
            }
            tabH[rem] = acc;
        }
    }
    __syncthreads();
    for (int idx = tid; idx < H_ * F_ * F_; idx += blockDim.x) {
        int h = idx / (F_ * F_), rem = idx % (F_ * F_);
        int i = rem / F_, j = rem % F_;
        bias_f[idx] = tabT[(i - j + F_ - 1) * H_ + h];
    }
    for (int idx = tid; idx < H_ * P_ * P_; idx += blockDim.x) {
        int h = idx / (P_ * P_), rem = idx % (P_ * P_);
        int p = rem / P_, q = rem % P_;
        bias_p[idx] = tabH[hops[p * P_ + q] * H_ + h];
    }
}

// ---------------------------------------------------------------- QKV GEMM
// (N*T, D) x (3D, D)^T -> scattered q/k (N,H,T,HD), vT (N,H,HD,T), v fp32.
// Block = 128 threads (4 waves); wave owns one 16x16 tile of a 16x64 block.
__global__ void qkv_gemm_kernel(const u16* __restrict__ x_bf, const u16* __restrict__ w_bf,
                                u16* __restrict__ q_bf, u16* __restrict__ k_bf,
                                u16* __restrict__ vT_bf, float* __restrict__ v_f32) {
    int wave = threadIdx.x >> 5, lane = threadIdx.x & 31;
    int half = lane >> 4, ln = lane & 15;
    int m0 = blockIdx.x * 16;
    int c  = blockIdx.y * 64 + wave * 16 + ln;
    const u16* arow = x_bf + (size_t)(m0 + ln) * D_;
    const u16* bcol = w_bf + (size_t)c * D_;
    v8f acc = {0.f, 0.f, 0.f, 0.f, 0.f, 0.f, 0.f, 0.f};
    for (int kb = 0; kb < D_; kb += 32) {
        v16bf a = load_a_frag(arow + kb, half);
        v16bf b = load_b_frag(bcol + kb, half);
        acc = __builtin_amdgcn_wmma_f32_16x16x32_bf16(false, a, false, b,
                                                      (short)0, acc, false, false);
    }
    int s   = c / D_;            // 0=q 1=k 2=v
    int rem = c - s * D_;
    int h = rem >> 5, d = rem & 31;
    #pragma unroll
    for (int r = 0; r < 8; r++) {
        int m = m0 + r + 8 * half;
        int n = m / T_;
        int t = m - n * T_;
        float val = acc[r];
        size_t idx = ((size_t)(n * H_ + h) * T_ + t) * HD_ + d;
        if (s == 0)      q_bf[idx] = f2bf(val);
        else if (s == 1) k_bf[idx] = f2bf(val);
        else {
            vT_bf[((size_t)(n * H_ + h) * HD_ + d) * T_ + t] = f2bf(val);
            v_f32[idx] = val;
        }
    }
}

// ---------------------------------------------------------------- flash attn
// One wave per (n, h, 16-row q tile).  Online softmax over 50 steps of 32 keys.
__global__ void flash_attn_kernel(const u16* __restrict__ q_bf, const u16* __restrict__ k_bf,
                                  const u16* __restrict__ vT_bf, const float* __restrict__ v_f32,
                                  const float* __restrict__ bias_f, const float* __restrict__ bias_p,
                                  const float* __restrict__ outer, const float* __restrict__ alpha_p,
                                  u16* __restrict__ attn_out) {
    int t0 = blockIdx.x * 16;
    int h  = blockIdx.y;
    int n  = blockIdx.z;
    int lane = threadIdx.x, half = lane >> 4, ln = lane & 15;

    __shared__ u16 ptile[16 * 32];   // P tile staging for C-layout -> A-layout

    const u16* qrow = q_bf + ((size_t)(n * H_ + h) * T_ + (t0 + ln)) * HD_;
    v16bf aq = load_a_frag(qrow, half);

    const u16* kbase  = k_bf  + (size_t)(n * H_ + h) * T_ * HD_;
    const u16* vTbase = vT_bf + (size_t)(n * H_ + h) * HD_ * T_;
    const float* bF = bias_f + h * F_ * F_;
    const float* bP = bias_p + h * P_ * P_;

    float mrow[8], lsum[8];
    int tF[8], tP[8];
    v8f o0 = {0.f, 0.f, 0.f, 0.f, 0.f, 0.f, 0.f, 0.f};
    v8f o1 = {0.f, 0.f, 0.f, 0.f, 0.f, 0.f, 0.f, 0.f};
    #pragma unroll
    for (int r = 0; r < 8; r++) {
        int t = t0 + r + 8 * half;
        tF[r] = t / P_;
        tP[r] = t - tF[r] * P_;
        mrow[r] = -3.0e38f;
        lsum[r] = 0.f;
    }

    for (int s0 = 0; s0 < T_; s0 += 32) {
        // scores: q(16x32) x k^T(32x32) -> two 16x16 fragments
        v16bf bk0 = load_b_frag(kbase + (size_t)(s0 + ln) * HD_, half);
        v16bf bk1 = load_b_frag(kbase + (size_t)(s0 + 16 + ln) * HD_, half);
        v8f z = {0.f, 0.f, 0.f, 0.f, 0.f, 0.f, 0.f, 0.f};
        v8f sc0 = __builtin_amdgcn_wmma_f32_16x16x32_bf16(false, aq, false, bk0,
                                                          (short)0, z, false, false);
        v8f sc1 = __builtin_amdgcn_wmma_f32_16x16x32_bf16(false, aq, false, bk1,
                                                          (short)0, z, false, false);

        int s0c = s0 + ln,       s1c = s0 + 16 + ln;
        int s0F = s0c / P_,      s0P = s0c - s0F * P_;
        int s1F = s1c / P_,      s1P = s1c - s1F * P_;

        float p0[8], p1[8];
        #pragma unroll
        for (int r = 0; r < 8; r++) {
            float l0 = (sc0[r] + bF[tF[r] * F_ + s0F] + bP[tP[r] * P_ + s0P]) * SCALE_;
            float l1 = (sc1[r] + bF[tF[r] * F_ + s1F] + bP[tP[r] * P_ + s1P]) * SCALE_;
            float rmax = fmaxf(l0, l1);
            #pragma unroll
            for (int off = 1; off < 16; off <<= 1)
                rmax = fmaxf(rmax, __shfl_xor(rmax, off, 32));
            float mnew = fmaxf(mrow[r], rmax);
            float corr = __expf(mrow[r] - mnew);
            float e0 = __expf(l0 - mnew);
            float e1 = __expf(l1 - mnew);
            float rs = e0 + e1;
            #pragma unroll
            for (int off = 1; off < 16; off <<= 1)
                rs += __shfl_xor(rs, off, 32);
            lsum[r] = lsum[r] * corr + rs;
            mrow[r] = mnew;
            o0[r] *= corr;
            o1[r] *= corr;
            p0[r] = e0;
            p1[r] = e1;
        }

        __syncthreads();                        // WAR on ptile vs prev iter reads
        #pragma unroll
        for (int r = 0; r < 8; r++) {
            int M = r + 8 * half;
            ptile[M * 32 + ln]      = f2bf(p0[r]);
            ptile[M * 32 + 16 + ln] = f2bf(p1[r]);
        }
        __syncthreads();

        // P as A-fragment (16x32) straight from LDS rows (ds_load_b128 x2)
        union { U4 q[2]; v16bf v; } ua;
        ua.q[0] = *reinterpret_cast<const U4*>(&ptile[ln * 32 + half * 8]);
        ua.q[1] = *reinterpret_cast<const U4*>(&ptile[ln * 32 + half * 8 + 16]);
        v16bf ap = ua.v;

        v16bf bv0 = load_b_frag(vTbase + (size_t)ln * T_ + s0, half);
        v16bf bv1 = load_b_frag(vTbase + (size_t)(16 + ln) * T_ + s0, half);
        o0 = __builtin_amdgcn_wmma_f32_16x16x32_bf16(false, ap, false, bv0,
                                                     (short)0, o0, false, false);
        o1 = __builtin_amdgcn_wmma_f32_16x16x32_bf16(false, ap, false, bv1,
                                                     (short)0, o1, false, false);
    }

    // epilogue: alpha * softmax@v + relational term, store bf16 (N,T,D)
    float alpha = alpha_p[0];
    const float* vbase = v_f32 + (size_t)(n * H_ + h) * T_ * HD_;
    const float* orow  = outer + h * P_ * P_;
    #pragma unroll
    for (int r = 0; r < 8; r++) {
        int t = t0 + r + 8 * half;
        float inv = alpha / lsum[r];
        const float* vf = vbase + (size_t)(tF[r] * P_) * HD_;
        const float* ow = orow + tP[r] * P_;
        float r0 = 0.f, r1 = 0.f;
        for (int q = 0; q < P_; q++) {
            float w = ow[q];
            r0 += w * vf[q * HD_ + ln];
            r1 += w * vf[q * HD_ + 16 + ln];
        }
        u16* dst = attn_out + (size_t)(n * T_ + t) * D_ + h * HD_;
        dst[ln]      = f2bf(o0[r] * inv + r0);
        dst[16 + ln] = f2bf(o1[r] * inv + r1);
    }
}

// ---------------------------------------------------------------- proj GEMM
__global__ void proj_gemm_kernel(const u16* __restrict__ a_bf, const u16* __restrict__ w_bf,
                                 const float* __restrict__ bias, float* __restrict__ out) {
    int wave = threadIdx.x >> 5, lane = threadIdx.x & 31;
    int half = lane >> 4, ln = lane & 15;
    int m0 = blockIdx.x * 16;
    int c  = blockIdx.y * 64 + wave * 16 + ln;
    const u16* arow = a_bf + (size_t)(m0 + ln) * D_;
    const u16* bcol = w_bf + (size_t)c * D_;
    v8f acc = {0.f, 0.f, 0.f, 0.f, 0.f, 0.f, 0.f, 0.f};
    for (int kb = 0; kb < D_; kb += 32) {
        v16bf a = load_a_frag(arow + kb, half);
        v16bf b = load_b_frag(bcol + kb, half);
        acc = __builtin_amdgcn_wmma_f32_16x16x32_bf16(false, a, false, b,
                                                      (short)0, acc, false, false);
    }
    float bv = bias[c];
    #pragma unroll
    for (int r = 0; r < 8; r++)
        out[(size_t)(m0 + r + 8 * half) * D_ + c] = acc[r] + bv;
}

// ---------------------------------------------------------------- launcher
extern "C" void kernel_launch(void* const* d_in, const int* in_sizes, int n_in,
                              void* d_out, int out_size, void* d_ws, size_t ws_size,
                              hipStream_t stream) {
    (void)in_sizes; (void)n_in; (void)out_size; (void)ws_size;
    const float* x      = (const float*)d_in[0];
    const float* qkv_w  = (const float*)d_in[1];
    const float* proj_w = (const float*)d_in[2];
    const float* proj_b = (const float*)d_in[3];
    const float* t_w1   = (const float*)d_in[4];
    const float* t_b1   = (const float*)d_in[5];
    const float* t_w2   = (const float*)d_in[6];
    const float* t_b2   = (const float*)d_in[7];
    const float* h_w1   = (const float*)d_in[8];
    const float* h_b1   = (const float*)d_in[9];
    const float* h_w2   = (const float*)d_in[10];
    const float* h_b2   = (const float*)d_in[11];
    const float* outer  = (const float*)d_in[12];
    const float* alpha  = (const float*)d_in[13];
    const int*   hops   = (const int*)d_in[14];
    float* out = (float*)d_out;

    constexpr size_t NTD  = (size_t)N_ * T_ * D_;      // 2,457,600
    constexpr size_t QKVW = (size_t)3 * D_ * D_;       // 110,592
    constexpr size_t PW   = (size_t)D_ * D_;           // 36,864
    constexpr size_t QKV1 = (size_t)N_ * H_ * T_ * HD_;// 2,457,600

    auto align256 = [](size_t v) { return (v + 255) & ~(size_t)255; };
    char* ws = (char*)d_ws;
    size_t off = 0;
    u16*   x_bf    = (u16*)(ws + off);  off = align256(off + NTD * 2);
    u16*   qkvw_bf = (u16*)(ws + off);  off = align256(off + QKVW * 2);
    u16*   projw_bf= (u16*)(ws + off);  off = align256(off + PW * 2);
    u16*   q_bf    = (u16*)(ws + off);  off = align256(off + QKV1 * 2);
    u16*   k_bf    = (u16*)(ws + off);  off = align256(off + QKV1 * 2);
    u16*   vT_bf   = (u16*)(ws + off);  off = align256(off + QKV1 * 2);
    float* v_f32   = (float*)(ws + off);off = align256(off + QKV1 * 4);
    u16*   attn_bf = (u16*)(ws + off);  off = align256(off + NTD * 2);
    float* bias_f  = (float*)(ws + off);off = align256(off + (size_t)H_ * F_ * F_ * 4);
    float* bias_p  = (float*)(ws + off);off = align256(off + (size_t)H_ * P_ * P_ * 4);

    f32_to_bf16_kernel<<<(int)((NTD  + 255) / 256), 256, 0, stream>>>(x, x_bf, (int)NTD);
    f32_to_bf16_kernel<<<(int)((QKVW + 255) / 256), 256, 0, stream>>>(qkv_w, qkvw_bf, (int)QKVW);
    f32_to_bf16_kernel<<<(int)((PW   + 255) / 256), 256, 0, stream>>>(proj_w, projw_bf, (int)PW);

    bias_tables_kernel<<<1, 256, 0, stream>>>(t_w1, t_b1, t_w2, t_b2,
                                              h_w1, h_b1, h_w2, h_b2,
                                              hops, bias_f, bias_p);

    qkv_gemm_kernel<<<dim3((N_ * T_) / 16, (3 * D_) / 64), 128, 0, stream>>>(
        x_bf, qkvw_bf, q_bf, k_bf, vT_bf, v_f32);

    flash_attn_kernel<<<dim3(T_ / 16, H_, N_), 32, 0, stream>>>(
        q_bf, k_bf, vT_bf, v_f32, bias_f, bias_p, outer, alpha, attn_bf);

    proj_gemm_kernel<<<dim3((N_ * T_) / 16, D_ / 64), 128, 0, stream>>>(
        attn_bf, projw_bf, proj_b, out);
}
